// CstrFlashModel_11879879544583
// MI455X (gfx1250) — matrix operand order repaired and checked by reading, not compiled, not run on hardware
//
#include <hip/hip_runtime.h>
#include <math.h>

// Model constants (from reference): Np=5, Ny=8, Nu=4, NZ=60, hidden=128
#define NP   5
#define NY   8
#define NU   4
#define NZK  60
#define ZPAD 64          // z K-dim padded to 64 for 2 x K=32 WMMA steps
#define HDIM 128
#define TT   1024
#define BBATCH 8192
#define MT   16          // batch rows per workgroup (one WMMA M tile)
#define NWAVE 4          // cooperating waves per tile
#define BLOCK (NWAVE*32)

typedef __attribute__((ext_vector_type(16))) _Float16 v16h;
typedef __attribute__((ext_vector_type(8)))  float    v8f;
typedef __attribute__((ext_vector_type(4)))  unsigned int u32x4;
typedef __attribute__((ext_vector_type(2)))  unsigned int u32x2;

union AFrag { v16h v; _Float16 h[16]; u32x4 q[2]; };

// ---- branch-free tanh: single v_tanh_f32 if available, else exp2/rcp ----
__device__ __forceinline__ float fast_tanh(float x) {
#if __has_builtin(__builtin_amdgcn_tanhf)
  return __builtin_amdgcn_tanhf(x);
#else
  // tanh(x) = 1 - 2/(exp2(2*log2(e)*x) + 1); exp2 over/underflow saturates to +-1
  float e = __builtin_amdgcn_exp2f(x * 2.8853900817779268f);
  return 1.0f - 2.0f * __builtin_amdgcn_rcpf(e + 1.0f);
#endif
}

// ---- WMMA wrapper: D = A(16x32 f16) x B(32x16 f16) + C(16x16 f32) ----
__device__ __forceinline__ v8f mma(v16h a, v16h b, v8f c) {
  return __builtin_amdgcn_wmma_f32_16x16x32_f16(false, a, false, b,
                                                (short)0, c, false, false);
}

// ---- B fragment from transposed-weight LDS (Wt[n][k], K contiguous) ----
// Lane layout: N = lane%16, K = (lane/16)*8 + i (i<8), 16 + (lane/16)*8 + (i-8) (i>=8)
__device__ __forceinline__ v16h wload(const _Float16* Wt, int kstride,
                                      int ntile, int kf, int lane) {
  AFrag f;
  const _Float16* p = Wt + (ntile * 16 + (lane & 15)) * kstride
                         + kf * 32 + ((lane >> 4) << 3);
  f.q[0] = *(const u32x4*)p;
  f.q[1] = *(const u32x4*)(p + 16);
  return f.v;
}

// ---- A fragment from an LDS activation row (row-major, 16-bit A layout) ----
// halves i=0..7  -> K = kf*32 + (lane/16)*8 + i
// halves i=8..15 -> K = kf*32 + 16 + (lane/16)*8 + (i-8)
__device__ __forceinline__ v16h hload(const _Float16* hrow, int kf, int lane) {
  AFrag f;
  int kb = kf * 32 + ((lane >> 4) << 3);
  f.q[0] = *(const u32x4*)(hrow + kb);
  f.q[1] = *(const u32x4*)(hrow + kb + 16);
  return f.v;
}

// ---- 8 logical-K halves of the ring-buffered z row, starting at s (mult of 8)
// physical row layout: y blocks p=0..4 at halves [p*8, p*8+8), u blocks at
// halves [40+p*4, 40+p*4+4). logical block j -> physical (head+j)%5.
__device__ __forceinline__ u32x4 ldgrp(const _Float16* zrow, int s, int head) {
  if (s < 40) {                       // one whole y block
    int p = head + (s >> 3); if (p >= NP) p -= NP;
    return *(const u32x4*)(zrow + (p << 3));
  } else if (s < 56) {                // two u blocks
    int b = (s - 40) >> 2;
    int p0 = head + b;     if (p0 >= NP) p0 -= NP;
    int p1 = p0 + 1;       if (p1 >= NP) p1 -= NP;
    u32x2 lo = *(const u32x2*)(zrow + 40 + (p0 << 2));
    u32x2 hi = *(const u32x2*)(zrow + 40 + (p1 << 2));
    u32x4 r; r.x = lo.x; r.y = lo.y; r.z = hi.x; r.w = hi.y;
    return r;
  } else {                            // last u block + zero pad (K 60..63)
    int p = head + 4; if (p >= NP) p -= NP;
    u32x2 lo = *(const u32x2*)(zrow + 40 + (p << 2));
    u32x4 r; r.x = lo.x; r.y = lo.y; r.z = 0u; r.w = 0u;
    return r;
  }
}

__device__ __forceinline__ v16h zload(const _Float16* zrow, int f, int head,
                                      int lane) {
  AFrag a;
  int off = (lane >> 4) << 3;
  a.q[0] = ldgrp(zrow, f * 32 + off,      head);
  a.q[1] = ldgrp(zrow, f * 32 + 16 + off, head);
  return a.v;
}

__global__ __launch_bounds__(BLOCK) void cstr_scan_kernel(
    const float* __restrict__ u,  const float* __restrict__ z0,
    const float* __restrict__ W1, const float* __restrict__ b1,
    const float* __restrict__ W2, const float* __restrict__ b2,
    const float* __restrict__ W3, const float* __restrict__ b3,
    float* __restrict__ out)
{
  __shared__ alignas(16) _Float16 sW1t[HDIM * ZPAD];   // [n][k] 16 KB
  __shared__ alignas(16) _Float16 sW2t[HDIM * HDIM];   // [n][k] 32 KB
  __shared__ alignas(16) _Float16 sW3t[16 * HDIM];     // [n][k] 4 KB (n>=8 zero)
  __shared__ float sb1[HDIM], sb2[HDIM], sb3[16];
  __shared__ alignas(16) _Float16 sZ [MT * ZPAD];      // ring-buffered state
  __shared__ alignas(16) _Float16 sH [MT * HDIM];      // layer-1 activations
  __shared__ alignas(16) _Float16 sH2[MT * HDIM];      // layer-2 activations

  const int tid   = threadIdx.x;
  const int lane  = tid & 31;
  const int wave  = tid >> 5;
  const int bbase = blockIdx.x * MT;

  // ---- one-time: convert weights f32->f16, transposed, into LDS ----
  for (int i = tid; i < HDIM * ZPAD; i += BLOCK) {
    int n = i >> 6, k = i & 63;
    sW1t[i] = (k < NZK) ? (_Float16)W1[k * HDIM + n] : (_Float16)0.f;
  }
  for (int i = tid; i < HDIM * HDIM; i += BLOCK) {
    int n = i >> 7, k = i & 127;
    sW2t[i] = (_Float16)W2[k * HDIM + n];
  }
  for (int i = tid; i < 16 * HDIM; i += BLOCK) {
    int n = i >> 7, k = i & 127;
    sW3t[i] = (n < NY) ? (_Float16)W3[k * NY + n] : (_Float16)0.f;
  }
  for (int i = tid; i < HDIM; i += BLOCK) { sb1[i] = b1[i]; sb2[i] = b2[i]; }
  if (tid < 16) sb3[tid] = (tid < NY) ? b3[tid] : 0.f;

  // ---- one-time: z0 into the ring (head=0 => logical == physical) ----
  for (int i = tid; i < MT * ZPAD; i += BLOCK) {
    int m = i >> 6, k = i & 63;
    sZ[i] = (k < NZK) ? (_Float16)z0[(size_t)(bbase + m) * NZK + k]
                      : (_Float16)0.f;
  }
  __syncthreads();

  const int nsel  = lane & 15;           // N within a tile / A-row id
  const int mhalf = (lane >> 4) << 3;    // C-layout M offset: 0 or 8
  const _Float16* zrow = sZ + nsel * ZPAD;
  const _Float16* hrow = sH + nsel * HDIM;
  const _Float16* h2row = sH2 + nsel * HDIM;
  const int nt0 = wave * 2;              // this wave's two N tiles

  // hoist the 4 W3 B-fragments (only wave 0 uses them)
  v16h w3f[4];
#pragma unroll
  for (int kf = 0; kf < 4; ++kf) w3f[kf] = wload(sW3t, HDIM, 0, kf, lane);
  const float bias3 = sb3[nsel];

  int head = 0;
  for (int t = 0; t < TT; ++t) {
    // wave 1: fetch u_t early (consumed only at end of step)
    float u0 = 0.f, u1 = 0.f, u2 = 0.f, u3 = 0.f;
    if (wave == 1 && lane < 16) {
      const float* up = u + ((size_t)(bbase + lane) * TT + t) * NU;
      u0 = up[0]; u1 = up[1]; u2 = up[2]; u3 = up[3];
      __builtin_prefetch(up + NU, 0, 0);
    }

    // ---- layer 1: h = tanh(z @ W1 + b1), this wave's 2 N tiles ----
    v16h za0 = zload(zrow, 0, head, lane);
    v16h za1 = zload(zrow, 1, head, lane);
#pragma unroll
    for (int j = 0; j < 2; ++j) {
      int n = nt0 + j;
      float bb = sb1[n * 16 + nsel];
      v8f c = {bb, bb, bb, bb, bb, bb, bb, bb};
      c = mma(za0, wload(sW1t, ZPAD, n, 0, lane), c);
      c = mma(za1, wload(sW1t, ZPAD, n, 1, lane), c);
#pragma unroll
      for (int v = 0; v < 8; ++v)
        sH[(mhalf + v) * HDIM + n * 16 + nsel] = (_Float16)fast_tanh(c[v]);
    }
    __syncthreads();

    // ---- layer 2: h2 = tanh(h @ W2 + b2) ----
    v16h ha[4];
#pragma unroll
    for (int kf = 0; kf < 4; ++kf) ha[kf] = hload(hrow, kf, lane);
#pragma unroll
    for (int j = 0; j < 2; ++j) {
      int n = nt0 + j;
      float bb = sb2[n * 16 + nsel];
      v8f c = {bb, bb, bb, bb, bb, bb, bb, bb};
#pragma unroll
      for (int kf = 0; kf < 4; ++kf)
        c = mma(ha[kf], wload(sW2t, HDIM, n, kf, lane), c);
#pragma unroll
      for (int v = 0; v < 8; ++v)
        sH2[(mhalf + v) * HDIM + n * 16 + nsel] = (_Float16)fast_tanh(c[v]);
    }
    __syncthreads();

    // ---- layer 3 (wave 0): y = h2 @ W3 + b3; emit + recycle into ring ----
    if (wave == 0) {
      v16h h2a[4];
#pragma unroll
      for (int kf = 0; kf < 4; ++kf) h2a[kf] = hload(h2row, kf, lane);
      v8f c = {bias3, bias3, bias3, bias3, bias3, bias3, bias3, bias3};
#pragma unroll
      for (int kf = 0; kf < 4; ++kf) c = mma(h2a[kf], w3f[kf], c);
      if (nsel < NY) {
#pragma unroll
        for (int v = 0; v < 8; ++v) {
          int m = mhalf + v;
          float yv = c[v];
          out[((size_t)(bbase + m) * TT + t) * NY + nsel] = yv;
          sZ[m * ZPAD + head * NY + nsel] = (_Float16)yv;  // newest y block
        }
      }
    }
    // wave 1: insert u_t into the ring (overlaps wave 0's layer 3)
    if (wave == 1 && lane < 16) {
      _Float16* zu = sZ + lane * ZPAD + 40 + head * NU;
      zu[0] = (_Float16)u0; zu[1] = (_Float16)u1;
      zu[2] = (_Float16)u2; zu[3] = (_Float16)u3;
    }
    __syncthreads();

    head = head + 1; if (head == NP) head = 0;   // rotate window
  }
}

extern "C" void kernel_launch(void* const* d_in, const int* in_sizes, int n_in,
                              void* d_out, int out_size, void* d_ws, size_t ws_size,
                              hipStream_t stream) {
  (void)in_sizes; (void)n_in; (void)out_size; (void)d_ws; (void)ws_size;
  const float* u  = (const float*)d_in[0];
  const float* z0 = (const float*)d_in[1];
  const float* W1 = (const float*)d_in[2];
  const float* b1 = (const float*)d_in[3];
  const float* W2 = (const float*)d_in[4];
  const float* b2 = (const float*)d_in[5];
  const float* W3 = (const float*)d_in[6];
  const float* b3 = (const float*)d_in[7];
  float* out = (float*)d_out;

  dim3 grid(BBATCH / MT);   // 512 workgroups, one 16-row batch tile each
  dim3 block(BLOCK);        // 4 wave32 per tile
  cstr_scan_kernel<<<grid, block, 0, stream>>>(u, z0, W1, b1, W2, b2, W3, b3, out);
}
